// AttentionDispatcher_67860483277088
// MI455X (gfx1250) — matrix operand order repaired and checked by reading, not compiled, not run on hardware
//
#include <hip/hip_runtime.h>
#include <hip/hip_bf16.h>

typedef __attribute__((ext_vector_type(16))) _Float16 v16h;
typedef __attribute__((ext_vector_type(8)))  _Float16 v8h;
typedef __attribute__((ext_vector_type(8)))  float    v8f;

#define BATCH 4
#define NROWS 4096
#define DMODEL 512
#define NHEAD 8
#define RS 1024
#define LDST 40   // LDS tile row stride in halves (80B, 16B-aligned, conflict-free)

__device__ __forceinline__ int laneid() { return threadIdx.x & 31; }

// A-fragment (16x32 f16): lane (m = lane&15, h = lane>>4) holds K = 16h+{0..15}
__device__ __forceinline__ v16h loadA(const _Float16* p, int s) {
    int lane = laneid(), m = lane & 15, h = lane >> 4;
    const _Float16* r = p + m * s + h * 16;
    v8h lo = *(const v8h*)r;
    v8h hi = *(const v8h*)(r + 8);
    return __builtin_shufflevector(lo, hi, 0, 1, 2, 3, 4, 5, 6, 7, 8, 9, 10, 11, 12, 13, 14, 15);
}

// B-fragment (32x16 f16) from column-major tile p[n][k]: lane col n = lane&15, K = 16h+{0..15}
__device__ __forceinline__ v16h loadB16(const _Float16* p, int s) {
    int lane = laneid(), n = lane & 15, h = lane >> 4;
    const _Float16* r = p + n * s + h * 16;
    v8h lo = *(const v8h*)r;
    v8h hi = *(const v8h*)(r + 8);
    return __builtin_shufflevector(lo, hi, 0, 1, 2, 3, 4, 5, 6, 7, 8, 9, 10, 11, 12, 13, 14, 15);
}

__device__ __forceinline__ v8f wmma_f16(v16h a, v16h b, v8f c) {
    return __builtin_amdgcn_wmma_f32_16x16x32_f16(false, a, false, b, (short)0, c, false, false);
}

__device__ __forceinline__ float halfmax(float v) {
#pragma unroll
    for (int m = 1; m < 16; m <<= 1) v = fmaxf(v, __shfl_xor(v, m, 16));
    return v;
}
__device__ __forceinline__ float halfsum(float v) {
#pragma unroll
    for (int m = 1; m < 16; m <<= 1) v += __shfl_xor(v, m, 16);
    return v;
}

// ------------- Kernel 0: one-time weight convert+transpose: Wt[m][n][k] = f16(W_m[k][n]) -------------
__global__ __launch_bounds__(256) void wtrans_kernel(
    const float* __restrict__ Wq, const float* __restrict__ Wk,
    const float* __restrict__ Wv, const float* __restrict__ Wo,
    _Float16* __restrict__ Wt) {
    const int m = blockIdx.y;
    const float* W = (m == 0) ? Wq : (m == 1) ? Wk : (m == 2) ? Wv : Wo;
    int id = blockIdx.x * 256 + threadIdx.x;     // 512 n * 32 k-chunks
    int n = id >> 5, kb = (id & 31) * 16;
    _Float16* dst = Wt + ((size_t)m * DMODEL + n) * DMODEL + kb;
#pragma unroll
    for (int i = 0; i < 16; ++i) dst[i] = (_Float16)W[(size_t)(kb + i) * DMODEL + n];
}

// ------------- Kernel 1: gathered projections; V written transposed (feat-major) -------------
// blockIdx.y encodes 11 needed (set, weight) pairs: y<3 -> Q sets 0..2; y<7 -> K sets 0..3; else V sets 0..3
__global__ __launch_bounds__(128) void proj_kernel(
    const float* __restrict__ x,
    const int* __restrict__ i0, const int* __restrict__ i1,
    const int* __restrict__ i2, const int* __restrict__ i3,
    const _Float16* __restrict__ Wt,
    _Float16* __restrict__ Qg, _Float16* __restrict__ Kg, _Float16* __restrict__ VgT) {
    const int tm = blockIdx.x >> 3, tn = blockIdx.x & 7;   // 128-row x 64-col macro tile
    const int y = blockIdx.y;
    const int wsel = (y < 3) ? 0 : (y < 7) ? 1 : 2;
    const int r    = (y < 3) ? y : (y < 7) ? (y - 3) : (y - 7);
    const int b = blockIdx.z;
    const int* idx = (r == 0) ? i0 : (r == 1) ? i1 : (r == 2) ? i2 : i3;
    const _Float16* Wm = Wt + (size_t)wsel * DMODEL * DMODEL;

    __shared__ __align__(16) _Float16 Xs[128][LDST];
    __shared__ __align__(16) _Float16 Ws[64][LDST];   // column-major: Ws[n][k]

    const int t = threadIdx.x, w = t >> 5;
    v8f acc[2][4] = {};

    const int gid = idx[tm * 128 + t];
    const float* xrow = x + ((size_t)b * NROWS + gid) * DMODEL;
    const int wn = t >> 1, wh = (t & 1) * 16;
    const _Float16* wrow = Wm + (size_t)(tn * 64 + wn) * DMODEL + wh;

    for (int kc = 0; kc < 16; ++kc) {
        {   // X tile: one row per thread, f32 -> f16
            const float* src = xrow + kc * 32;
#pragma unroll
            for (int c = 0; c < 4; ++c) {
                v8h tmp;
#pragma unroll
                for (int i = 0; i < 8; ++i) tmp[i] = (_Float16)src[c * 8 + i];
                *(v8h*)&Xs[t][c * 8] = tmp;
            }
        }
        {   // W tile: already f16 column-major in global, pure b128 copies
            const _Float16* src = wrow + kc * 32;
            *(v8h*)&Ws[wn][wh]     = *(const v8h*)src;
            *(v8h*)&Ws[wn][wh + 8] = *(const v8h*)(src + 8);
        }
        __syncthreads();
        v16h a0 = loadA(&Xs[w * 32][0], LDST);
        v16h a1 = loadA(&Xs[w * 32 + 16][0], LDST);
#pragma unroll
        for (int nc = 0; nc < 4; ++nc) {
            v16h bf = loadB16(&Ws[nc * 16][0], LDST);
            acc[0][nc] = wmma_f16(a0, bf, acc[0][nc]);
            acc[1][nc] = wmma_f16(a1, bf, acc[1][nc]);
        }
        __syncthreads();
    }

    const int lane = laneid(), n = lane & 15, h = lane >> 4;
    const int row00 = tm * 128 + w * 32;
    const int col0 = tn * 64 + n;
    if (wsel == 2) {
        // V: feat-major layout VgT[bset][col][row]
        _Float16* base = VgT + (size_t)(b * 4 + r) * DMODEL * RS;
#pragma unroll
        for (int rb = 0; rb < 2; ++rb)
#pragma unroll
            for (int nc = 0; nc < 4; ++nc)
#pragma unroll
                for (int j = 0; j < 8; ++j)
                    base[(size_t)(col0 + nc * 16) * RS + row00 + rb * 16 + j + 8 * h] =
                        (_Float16)acc[rb][nc][j];
    } else {
        _Float16* base = (wsel == 0 ? Qg : Kg) + (size_t)(b * 4 + r) * RS * DMODEL;
#pragma unroll
        for (int rb = 0; rb < 2; ++rb)
#pragma unroll
            for (int nc = 0; nc < 4; ++nc)
#pragma unroll
                for (int j = 0; j < 8; ++j)
                    base[(size_t)(row00 + rb * 16 + j + 8 * h) * DMODEL + col0 + nc * 16] =
                        (_Float16)acc[rb][nc][j];
    }
}

// ------------- Kernel 2: flash attention per (qtile64, op, head, batch) -------------
__global__ __launch_bounds__(128) void attn_kernel(
    const _Float16* __restrict__ Qg, const _Float16* __restrict__ Kg,
    const _Float16* __restrict__ VgT, _Float16* __restrict__ Og) {
    const int S_OF[4] = {0, 0, 1, 2};
    const int D_OF[4] = {1, 2, 3, 0};
    const int qt = blockIdx.x;
    const int op = blockIdx.y >> 3, head = blockIdx.y & 7;
    const int b = blockIdx.z;
    const int bs = b * 4 + S_OF[op], bd = b * 4 + D_OF[op];
    const int t = threadIdx.x, w = t >> 5;
    const int lane = laneid(), n = lane & 15, h = lane >> 4;

    const _Float16* Qb = Qg + ((size_t)bs * RS + qt * 64 + w * 16) * DMODEL + head * 64;
    const _Float16* Kb = Kg + (size_t)bd * RS * DMODEL + head * 64;
    const _Float16* Vb = VgT + ((size_t)bd * DMODEL + head * 64) * RS;

    v16h qa0 = loadA(Qb, DMODEL);
    v16h qa1 = loadA(Qb + 32, DMODEL);

    __shared__ __align__(16) _Float16 Kt[32][LDST];     // [key][feat]
    __shared__ __align__(16) _Float16 VtT[64][LDST];    // [feat][key]
    __shared__ __align__(16) _Float16 Pw[4][16][LDST];

    v8f acc[4] = {};
    float mrow[8], lrow[8];
#pragma unroll
    for (int j = 0; j < 8; ++j) { mrow[j] = -3.0e38f; lrow[j] = 0.0f; }

    const int krow = t >> 2, kcb = (t & 3) * 16;
    const int vfr = t >> 1, vkh = (t & 1) * 16;

    for (int kt = 0; kt < 32; ++kt) {
        {   // stage K (row-major) and V (feat-major) tiles: pure b128 traffic
            const _Float16* ks = Kb + (size_t)(kt * 32 + krow) * DMODEL + kcb;
            *(v8h*)&Kt[krow][kcb]     = *(const v8h*)ks;
            *(v8h*)&Kt[krow][kcb + 8] = *(const v8h*)(ks + 8);
            const _Float16* vs = Vb + (size_t)vfr * RS + kt * 32 + vkh;
            *(v8h*)&VtT[vfr][vkh]     = *(const v8h*)vs;
            *(v8h*)&VtT[vfr][vkh + 8] = *(const v8h*)(vs + 8);
        }
        __syncthreads();

        v8f s0 = {}, s1 = {};
        s0 = wmma_f16(qa0, loadB16(&Kt[0][0], LDST), s0);
        s0 = wmma_f16(qa1, loadB16(&Kt[0][32], LDST), s0);
        s1 = wmma_f16(qa0, loadB16(&Kt[16][0], LDST), s1);
        s1 = wmma_f16(qa1, loadB16(&Kt[16][32], LDST), s1);

        float p0[8], p1[8], alpha[8];
#pragma unroll
        for (int j = 0; j < 8; ++j) {
            float v0 = s0[j] * 0.125f;   // 1/sqrt(64)
            float v1 = s1[j] * 0.125f;
            float mt = halfmax(fmaxf(v0, v1));
            float mn = fmaxf(mrow[j], mt);
            alpha[j] = __expf(mrow[j] - mn);
            mrow[j] = mn;
            p0[j] = __expf(v0 - mn);
            p1[j] = __expf(v1 - mn);
            lrow[j] = lrow[j] * alpha[j] + halfsum(p0[j] + p1[j]);
        }
#pragma unroll
        for (int nc = 0; nc < 4; ++nc)
#pragma unroll
            for (int j = 0; j < 8; ++j) acc[nc][j] *= alpha[j];

#pragma unroll
        for (int j = 0; j < 8; ++j) {
            Pw[w][j + 8 * h][n]      = (_Float16)p0[j];
            Pw[w][j + 8 * h][16 + n] = (_Float16)p1[j];
        }
        __syncthreads();

        v16h pa = loadA(&Pw[w][0][0], LDST);
#pragma unroll
        for (int nc = 0; nc < 4; ++nc)
            acc[nc] = wmma_f16(pa, loadB16(&VtT[nc * 16][0], LDST), acc[nc]);
        __syncthreads();
    }

    _Float16* obase = Og + ((size_t)(op * BATCH + b) * RS + qt * 64 + w * 16) * DMODEL + head * 64;
#pragma unroll
    for (int j = 0; j < 8; ++j) {
        float inv = 1.0f / lrow[j];
#pragma unroll
        for (int nc = 0; nc < 4; ++nc)
            obase[(size_t)(j + 8 * h) * DMODEL + nc * 16 + n] = (_Float16)(acc[nc][j] * inv);
    }
}

// ------------- Kernel 3: output projection with fused op-combine + scatter -------------
__global__ __launch_bounds__(128) void outproj_kernel(
    const _Float16* __restrict__ Og, const _Float16* __restrict__ Wt,
    const int* __restrict__ i0, const int* __restrict__ i1, const int* __restrict__ i2,
    float* __restrict__ out) {
    const int tm = blockIdx.x >> 3, tn = blockIdx.x & 7;   // 128 x 64 macro tile
    const int r = blockIdx.y;                               // 0..2
    const int b = blockIdx.z;
    const int* idx = (r == 0) ? i0 : (r == 1) ? i1 : i2;
    const int opA = (r == 0) ? 0 : (r == 1) ? 2 : 3;
    const bool dual = (r == 0);
    const _Float16* Wm = Wt + (size_t)3 * DMODEL * DMODEL;  // Wo

    __shared__ __align__(16) _Float16 Zs[128][LDST];
    __shared__ __align__(16) _Float16 Ws[64][LDST];

    const int t = threadIdx.x, w = t >> 5;
    v8f acc[2][4] = {};

    const size_t rowbase = ((size_t)(opA * BATCH + b) * RS + tm * 128 + t) * DMODEL;
    const int wn = t >> 1, wh = (t & 1) * 16;
    const _Float16* wrow = Wm + (size_t)(tn * 64 + wn) * DMODEL + wh;

    for (int kc = 0; kc < 16; ++kc) {
        {   // Z tile: combined attention outputs
            const _Float16* sA = Og + rowbase + kc * 32;
            const _Float16* sB = sA + (size_t)BATCH * RS * DMODEL;  // op1 follows op0
#pragma unroll
            for (int c = 0; c < 4; ++c) {
                v8h a = *(const v8h*)(sA + c * 8);
                v8h z;
                if (dual) {
                    v8h bb = *(const v8h*)(sB + c * 8);
#pragma unroll
                    for (int i = 0; i < 8; ++i)
                        z[i] = (_Float16)(((float)a[i] + 0.5f * (float)bb[i]) * (1.0f / 1.5f));
                } else {
                    z = a;
                }
                *(v8h*)&Zs[t][c * 8] = z;
            }
        }
        {
            const _Float16* src = wrow + kc * 32;
            *(v8h*)&Ws[wn][wh]     = *(const v8h*)src;
            *(v8h*)&Ws[wn][wh + 8] = *(const v8h*)(src + 8);
        }
        __syncthreads();
        v16h a0 = loadA(&Zs[w * 32][0], LDST);
        v16h a1 = loadA(&Zs[w * 32 + 16][0], LDST);
#pragma unroll
        for (int nc = 0; nc < 4; ++nc) {
            v16h bf = loadB16(&Ws[nc * 16][0], LDST);
            acc[0][nc] = wmma_f16(a0, bf, acc[0][nc]);
            acc[1][nc] = wmma_f16(a1, bf, acc[1][nc]);
        }
        __syncthreads();
    }

    const int lane = laneid(), n = lane & 15, h = lane >> 4;
#pragma unroll
    for (int rb = 0; rb < 2; ++rb)
#pragma unroll
        for (int j = 0; j < 8; ++j) {
            int gid = idx[tm * 128 + w * 32 + rb * 16 + j + 8 * h];
            float* dst = out + ((size_t)b * NROWS + gid) * DMODEL + tn * 64;
#pragma unroll
            for (int nc = 0; nc < 4; ++nc) dst[nc * 16 + n] = acc[rb][nc][j];
        }
}

// ------------- Kernel 4: pass-through rows of set 3 -------------
__global__ __launch_bounds__(256) void copy3_kernel(
    const float* __restrict__ x, const int* __restrict__ i3, float* __restrict__ out) {
    int row = blockIdx.x, b = blockIdx.y;
    int gid = i3[row];
    const float* s = x + ((size_t)b * NROWS + gid) * DMODEL;
    float* d = out + ((size_t)b * NROWS + gid) * DMODEL;
    for (int i = threadIdx.x; i < DMODEL; i += blockDim.x) d[i] = s[i];
}

extern "C" void kernel_launch(void* const* d_in, const int* in_sizes, int n_in,
                              void* d_out, int out_size, void* d_ws, size_t ws_size,
                              hipStream_t stream) {
    (void)in_sizes; (void)n_in; (void)out_size; (void)ws_size;
    const float* x  = (const float*)d_in[0];
    const int*   i0 = (const int*)d_in[1];
    const int*   i1 = (const int*)d_in[2];
    const int*   i2 = (const int*)d_in[3];
    const int*   i3 = (const int*)d_in[4];
    const float* Wq = (const float*)d_in[5];
    const float* Wk = (const float*)d_in[6];
    const float* Wv = (const float*)d_in[7];
    const float* Wo = (const float*)d_in[8];
    float* out = (float*)d_out;

    const size_t BUF = (size_t)BATCH * 4 * RS * DMODEL;  // 8M f16 elems = 16 MB
    _Float16* Qg  = (_Float16*)d_ws;
    _Float16* Kg  = Qg + BUF;
    _Float16* VgT = Kg + BUF;
    _Float16* Og  = VgT + BUF;
    _Float16* Wt  = Og + BUF;                            // 4*512*512 f16 = 2 MB

    wtrans_kernel<<<dim3(64, 4), 256, 0, stream>>>(Wq, Wk, Wv, Wo, Wt);

    dim3 g1(64, 11, BATCH);
    proj_kernel<<<g1, 128, 0, stream>>>(x, i0, i1, i2, i3, Wt, Qg, Kg, VgT);

    dim3 g2(RS / 64, 4 * NHEAD, BATCH);
    attn_kernel<<<g2, 128, 0, stream>>>(Qg, Kg, VgT, Og);

    dim3 g3(64, 3, BATCH);
    outproj_kernel<<<g3, 128, 0, stream>>>(Og, Wt, i0, i1, i2, out);

    copy3_kernel<<<dim3(RS, BATCH), 256, 0, stream>>>(x, i3, out);
}
